// ContrastiveLoss_57844619542578
// MI455X (gfx1250) — matrix operand order repaired and checked by reading, not compiled, not run on hardware
//
#include <hip/hip_runtime.h>
#include <hip/hip_bf16.h>

typedef __attribute__((ext_vector_type(16))) _Float16 v16h;
typedef __attribute__((ext_vector_type(8)))  _Float16 v8h;
typedef __attribute__((ext_vector_type(8)))  float    v8f;

#define B_ROWS   4096
#define D_DIM    256
#define NROWS    8192            // 2B
#define NTILES   512             // 8192 / 16
#define INV_T    2.0f            // 1 / TEMPERATURE (T = 0.5)
#define LDS_STRIDE 264           // halves per LDS tile row: 256 + 8 pad (16B)
                                 // row stride = 528 B = 132 dwords -> 4-bank
                                 // advance per row: conflict-free b128 reads

// ---------------------------------------------------------------------------
// Kernel 0: zero the rowsum accumulators and the output scalar.
// ---------------------------------------------------------------------------
__global__ __launch_bounds__(256) void ntxent_init(float* __restrict__ rowsum,
                                                   float* __restrict__ out) {
    int i = blockIdx.x * blockDim.x + threadIdx.x;
    if (i < NROWS) rowsum[i] = 0.0f;
    if (i == 0)    out[0]    = 0.0f;
}

// ---------------------------------------------------------------------------
// Kernel 1: L2-normalize both projections into f16 z[8192][256], and compute
// the positive cosine pos[i] = <z_i, z_{B+i}> in full f32.
// One wave per pair index i (8 waves / block).
// ---------------------------------------------------------------------------
__global__ __launch_bounds__(256) void ntxent_normalize(
        const float* __restrict__ p1, const float* __restrict__ p2,
        _Float16* __restrict__ z, float* __restrict__ pos) {
    const int lane = threadIdx.x & 31;
    const int wave = threadIdx.x >> 5;
    const int i    = blockIdx.x * 8 + wave;
    if (i >= B_ROWS) return;

    const float4* a4 = (const float4*)(p1 + (size_t)i * D_DIM + lane * 8);
    const float4* b4 = (const float4*)(p2 + (size_t)i * D_DIM + lane * 8);
    float4 a0 = a4[0], a1 = a4[1];
    float4 b0 = b4[0], b1 = b4[1];

    float ss1 = a0.x*a0.x + a0.y*a0.y + a0.z*a0.z + a0.w*a0.w
              + a1.x*a1.x + a1.y*a1.y + a1.z*a1.z + a1.w*a1.w;
    float ss2 = b0.x*b0.x + b0.y*b0.y + b0.z*b0.z + b0.w*b0.w
              + b1.x*b1.x + b1.y*b1.y + b1.z*b1.z + b1.w*b1.w;
    float dot = a0.x*b0.x + a0.y*b0.y + a0.z*b0.z + a0.w*b0.w
              + a1.x*b1.x + a1.y*b1.y + a1.z*b1.z + a1.w*b1.w;

    #pragma unroll
    for (int m = 16; m >= 1; m >>= 1) {        // full wave32 butterfly
        ss1 += __shfl_xor(ss1, m, 32);
        ss2 += __shfl_xor(ss2, m, 32);
        dot += __shfl_xor(dot, m, 32);
    }
    float inv1 = rsqrtf(ss1);
    float inv2 = rsqrtf(ss2);
    if (lane == 0) pos[i] = dot * inv1 * inv2;

    v8h h1, h2;
    float va[8] = {a0.x,a0.y,a0.z,a0.w,a1.x,a1.y,a1.z,a1.w};
    float vb[8] = {b0.x,b0.y,b0.z,b0.w,b1.x,b1.y,b1.z,b1.w};
    #pragma unroll
    for (int t = 0; t < 8; ++t) {
        h1[t] = (_Float16)(va[t] * inv1);
        h2[t] = (_Float16)(vb[t] * inv2);
    }
    *(v8h*)(z + (size_t)i * D_DIM + lane * 8)            = h1;
    *(v8h*)(z + (size_t)(i + B_ROWS) * D_DIM + lane * 8) = h2;
}

// ---------------------------------------------------------------------------
// Kernel 2: fused sim-GEMM + exp + diagonal mask + row reduction.
//
// blockIdx = rowGroup*8 + split. Wave w owns row tile I = rowGroup*8 + w
// (A fragments resident in VGPRs). All 8 waves sweep column tiles
// J = split + 8t cooperatively: each 16x256 f16 B tile (8 KB) is loaded
// ONCE per block into a double-buffered LDS stage (global loads for tile
// t+1 are issued before the WMMA chain of tile t to overlap), then all 8
// waves build their B fragments from LDS -> 8x less L2 traffic than
// per-wave streaming.
// ---------------------------------------------------------------------------
__global__ __launch_bounds__(256) void ntxent_simgemm(
        const _Float16* __restrict__ z, float* __restrict__ rowsum) {
    __shared__ _Float16 bbuf[2][16 * LDS_STRIDE];   // 2 x 8448 B

    const int tid     = threadIdx.x;
    const int lane    = tid & 31;
    const int wave    = tid >> 5;
    const int rowGrp  = blockIdx.x >> 3;
    const int split   = blockIdx.x & 7;
    const int I       = rowGrp * 8 + wave;          // this wave's row tile
    const int rowBase = I * 16;
    const int r0      = rowBase + (lane & 15);
    const int koffA   = (lane >> 4) * 8;            // A: half-wave K split of 8
    const int koffB   = (lane >> 4) * 8;            // B: half-wave K split of 8

    // ---- resident A fragments: 8 k-steps x v16h (64 VGPRs) ----
    v16h a[8];
    #pragma unroll
    for (int ks = 0; ks < 8; ++ks) {
        const _Float16* p = z + (size_t)r0 * D_DIM + ks * 32 + koffA;
        v8h lo = *(const v8h*)(p);        // K = koffA + 0..7
        v8h hi = *(const v8h*)(p + 16);   // K = koffA + 16..23
        a[ks] = __builtin_shufflevector(lo, hi,
                 0,1,2,3,4,5,6,7,8,9,10,11,12,13,14,15);
    }

    // ---- cooperative B-tile loader mapping: 256 thr x 32 B = 8 KB ----
    const int ldRow = tid >> 4;             // 0..15 tile row
    const int ldCol = (tid & 15) * 16;      // halves: 0..240 step 16

    // prologue: stage tile t=0
    {
        const _Float16* src = z + (size_t)(split * 16 + ldRow) * D_DIM + ldCol;
        v8h s0 = *(const v8h*)(src);
        v8h s1 = *(const v8h*)(src + 8);
        _Float16* dst = &bbuf[0][ldRow * LDS_STRIDE + ldCol];
        *(v8h*)(dst)     = s0;
        *(v8h*)(dst + 8) = s1;
    }
    __syncthreads();

    float acc[8] = {0.f,0.f,0.f,0.f,0.f,0.f,0.f,0.f};

    for (int t = 0; t < 64; ++t) {
        const int s = t & 1;
        const int J = split + (t << 3);

        // issue global loads for tile t+1 early (overlap with WMMA chain)
        v8h n0, n1;
        const bool hasNext = (t + 1 < 64);
        if (hasNext) {
            const int Jn = split + ((t + 1) << 3);
            const _Float16* src = z + (size_t)(Jn * 16 + ldRow) * D_DIM + ldCol;
            n0 = *(const v8h*)(src);
            n1 = *(const v8h*)(src + 8);
            if (t + 2 < 64) {               // warm L2 for tile t+2
                const int Jp = split + ((t + 2) << 3);
                __builtin_prefetch(z + (size_t)(Jp * 16 + ldRow) * D_DIM, 0, 3);
            }
        }

        // ---- WMMA chain: B fragments straight from LDS ----
        v8f c = {};
        #pragma unroll
        for (int ks = 0; ks < 8; ++ks) {
            const _Float16* pb =
                &bbuf[s][(lane & 15) * LDS_STRIDE + ks * 32 + koffB];
            v8h blo = *(const v8h*)(pb);
            v8h bhi = *(const v8h*)(pb + 16);
            v16h b = __builtin_shufflevector(blo, bhi,
                      0,1,2,3,4,5,6,7,8,9,10,11,12,13,14,15);
            c = __builtin_amdgcn_wmma_f32_16x16x32_f16(
                    false, a[ks], false, b, (short)0, c, false, false);
        }

        const bool diagTile = (J == I);
        #pragma unroll
        for (int r = 0; r < 8; ++r) {
            float e = __expf(c[r] * INV_T);
            if (diagTile && ((lane == r) || (lane == r + 24))) e = 0.0f;
            acc[r] += e;
        }

        __syncthreads();                    // all waves done reading bbuf
        if (hasNext) {
            _Float16* dst = &bbuf[s ^ 1][ldRow * LDS_STRIDE + ldCol];
            *(v8h*)(dst)     = n0;
            *(v8h*)(dst + 8) = n1;
        }
        __syncthreads();                    // staged tile visible to all
    }

    // ---- per-row reduction: butterfly within each 16-lane half ----
    #pragma unroll
    for (int r = 0; r < 8; ++r) {
        float v = acc[r];
        v += __shfl_xor(v, 1, 32);
        v += __shfl_xor(v, 2, 32);
        v += __shfl_xor(v, 4, 32);
        v += __shfl_xor(v, 8, 32);
        if ((lane & 15) == 0) {
            const int m = rowBase + r + ((lane >> 4) << 3);  // M = r or r+8
            atomicAdd(&rowsum[m], v);       // 8 split-blocks combine per row
        }
    }
}

// ---------------------------------------------------------------------------
// Kernel 3: loss = mean_i( log(denom_i) - pos_{i mod B} / T )
// ---------------------------------------------------------------------------
__global__ __launch_bounds__(256) void ntxent_finalize(
        const float* __restrict__ rowsum, const float* __restrict__ pos,
        float* __restrict__ out) {
    __shared__ float sdata[256];
    const int i = blockIdx.x * 256 + threadIdx.x;
    float term = 0.0f;
    if (i < NROWS) {
        const float p = pos[i & (B_ROWS - 1)];   // row i and row i+B share pos
        term = (__logf(rowsum[i]) - INV_T * p) * (1.0f / (float)NROWS);
    }
    sdata[threadIdx.x] = term;
    __syncthreads();
    #pragma unroll
    for (int s = 128; s > 0; s >>= 1) {
        if (threadIdx.x < s) sdata[threadIdx.x] += sdata[threadIdx.x + s];
        __syncthreads();
    }
    if (threadIdx.x == 0) atomicAdd(out, sdata[0]);
}

// ---------------------------------------------------------------------------
extern "C" void kernel_launch(void* const* d_in, const int* in_sizes, int n_in,
                              void* d_out, int out_size, void* d_ws, size_t ws_size,
                              hipStream_t stream) {
    const float* p1 = (const float*)d_in[0];   // proj_1 [4096,256] f32
    const float* p2 = (const float*)d_in[1];   // proj_2 [4096,256] f32
    float* out = (float*)d_out;                // scalar loss (f32)

    // workspace layout: z(f16 8192x256 = 4MB) | rowsum(8192 f32) | pos(4096 f32)
    _Float16* z      = (_Float16*)d_ws;
    float*    rowsum = (float*)((char*)d_ws + (size_t)NROWS * D_DIM * sizeof(_Float16));
    float*    pos    = rowsum + NROWS;

    ntxent_init     <<<(NROWS + 255) / 256, 256, 0, stream>>>(rowsum, out);
    ntxent_normalize<<<B_ROWS / 8,          256, 0, stream>>>(p1, p2, z, pos);
    ntxent_simgemm  <<<NTILES,              256, 0, stream>>>(z, rowsum);
    ntxent_finalize <<<(NROWS + 255) / 256, 256, 0, stream>>>(rowsum, pos, out);
}
